// SpatialMambaBlock_18476949307581
// MI455X (gfx1250) — matrix operand order repaired and checked by reading, not compiled, not run on hardware
//
#include <hip/hip_runtime.h>
#include <hip/hip_bf16.h>
#include <math.h>

// ---------------- shapes ----------------
#define SEQ    1944          // 8*243 sequences
#define LSEQ   17            // sequence length
#define CMOD   256           // d_model
#define DIN    512           // d_inner
#define DSTATE 128
#define NH     16            // heads
#define HD     32            // head dim
#define CONVD  768           // conv_dim = DIN + 2*DSTATE
#define DPROJ  1296          // 2*DIN + 2*DSTATE + NH
#define HID    1024          // hidden
#define TOK    (SEQ*LSEQ)    // 33048 tokens

typedef __attribute__((ext_vector_type(16))) _Float16 v16h;
typedef __attribute__((ext_vector_type(8)))  float    v8f;

// ---------------- WMMA helpers (wave32, ISA 7.12.2 layouts) ----------------
__device__ __forceinline__ v8f wmma_f16(v16h a, v16h b, v8f c) {
  return __builtin_amdgcn_wmma_f32_16x16x32_f16(false, a, false, b, (short)0, c,
                                                false, false);
}

// Fragment-packed storage: [tile][lane][16 halfs] -> one 32B read per lane.
__device__ __forceinline__ v16h frag16(const _Float16* p, int tile) {
  return ((const v16h*)p)[(size_t)tile * 32 + (threadIdx.x & 31)];
}

// Inverse maps used when *writing* packed fragments:
// A matrix 16x32 f16 (per ISA): lane = row | (((k>>3)&1)<<4), elem = (k&7)|((k>>4)<<3)
__device__ __forceinline__ int a_pack_idx(int tile, int row, int k) {
  int lane = (row & 15) | (((k >> 3) & 1) << 4);
  int e = (k & 7) | ((k >> 4) << 3);
  return (tile * 32 + lane) * 16 + e;
}
// B matrix 32x16 f16: lane = col | (((k>>3)&1)<<4), elem = (k&7)|((k>>4)<<3)

// C/D 16x16 f32: lanes 0-15 -> N=lane, rows v ; lanes 16-31 -> N=lane-16, rows v+8
__device__ __forceinline__ void store_acc_lds(float* s, int ld, int row0, int n0,
                                              v8f acc, int rowmax) {
  int lane = threadIdx.x & 31;
  int n = n0 + (lane & 15);
  int r0 = row0 + ((lane >> 4) << 3);
#pragma unroll
  for (int v = 0; v < 8; ++v) {
    int r = r0 + v;
    if (r < rowmax) s[r * ld + n] = acc[v];
  }
}

__device__ __forceinline__ float silu_f(float z) {
  return z / (1.f + __expf(-z));
}

// ------- weight pack: src[N][K] f32 -> fragment-packed f16 B tiles -------
// dst tile order: (nt * ktiles + kk), within tile [lane][16]
__global__ __launch_bounds__(256) void k_wpack(const float* __restrict__ src,
                                               _Float16* __restrict__ dst,
                                               int K, int N) {
  int i = blockIdx.x * 256 + threadIdx.x;
  if (i >= K * N) return;
  int k = i / N, n = i % N;
  int ktiles = K >> 5;
  int kk = k >> 5, kp = k & 31;
  int nt = n >> 4, np = n & 15;
  int lane = np | (((kp >> 3) & 1) << 4);
  int e = (kp & 7) | ((kp >> 4) << 3);
  dst[((size_t)(nt * ktiles + kk) * 32 + lane) * 16 + e] =
      (_Float16)src[(size_t)n * K + k];
}

// ---------------- LN1 + channel flip ----------------
__global__ __launch_bounds__(256) void k_ln1(const float* __restrict__ x,
                                             const float* __restrict__ w,
                                             const float* __restrict__ b,
                                             _Float16* __restrict__ xn,
                                             _Float16* __restrict__ xf) {
  int t = blockIdx.x;
  int c = threadIdx.x;
  float v = x[(size_t)t * CMOD + c];
  float s1 = v, s2 = v * v;
#pragma unroll
  for (int o = 16; o > 0; o >>= 1) {
    s1 += __shfl_xor(s1, o, 32);
    s2 += __shfl_xor(s2, o, 32);
  }
  __shared__ float r1[8], r2[8], mS, rsS;
  int wave = c >> 5, lane = c & 31;
  if (lane == 0) { r1[wave] = s1; r2[wave] = s2; }
  __syncthreads();
  if (c == 0) {
    float a1 = 0.f, a2 = 0.f;
    for (int i = 0; i < 8; ++i) { a1 += r1[i]; a2 += r2[i]; }
    float m = a1 / CMOD;
    mS = m;
    rsS = rsqrtf(a2 / CMOD - m * m + 1e-5f);
  }
  __syncthreads();
  float o = (v - mS) * rsS * w[c] + b[c];
  xn[(size_t)t * CMOD + c] = (_Float16)o;
  xf[(size_t)t * CMOD + (CMOD - 1 - c)] = (_Float16)o;
}

// ---------------- fused per-sequence Mamba branch ----------------
// LDS plan (dynamic):
//  sAp  : packed A frags, tiles (rt*8+kk), rt in {0,1}: 2*8*32*16 halfs (16 KB)
//  zx   : 17 x 1304 f32 (in-proj result; conv done in place)
//  Mb   : 17*17*16 f32 ; yb : 17*512 f32 ; dtv/cums : 17*16 ; CBs : 17*17
#define MAM_ZLD 1304
#define MAM_SA_HALFS (2 * 8 * 32 * 16)
#define MAM_ZX_F (LSEQ * MAM_ZLD)
#define MAM_MB_F (LSEQ * LSEQ * NH)
#define MAM_YB_F (LSEQ * DIN)
#define MAM_DT_F (LSEQ * NH)
#define MAM_CB_F (LSEQ * LSEQ)
#define MAM_SMEM_BYTES (MAM_SA_HALFS * 2 + \
  (MAM_ZX_F + MAM_MB_F + MAM_YB_F + 2 * MAM_DT_F + MAM_CB_F) * 4)

__global__ __launch_bounds__(256, 1) void k_mamba(
    const _Float16* __restrict__ xin16,  // [TOK,256]
    const _Float16* __restrict__ Winp,   // packed B, ktiles=8, ntiles=81
    const float* __restrict__ cw, const float* __restrict__ cb,
    const float* __restrict__ dtb, const float* __restrict__ Alog,
    const float* __restrict__ Dp, const float* __restrict__ nw,
    _Float16* __restrict__ yout)         // [TOK,512]
{
  extern __shared__ char smem[];
  _Float16* sAp = (_Float16*)smem;
  float* zx   = (float*)(smem + MAM_SA_HALFS * 2);
  float* Mb   = zx + MAM_ZX_F;
  float* yb   = Mb + MAM_MB_F;
  float* dtv  = yb + MAM_YB_F;
  float* cums = dtv + MAM_DT_F;
  float* CBs  = cums + MAM_DT_F;

  const int seq = blockIdx.x;
  const int tid = threadIdx.x;
  const int wave = tid >> 5;
  const int lane = tid & 31;
  const size_t tokbase = (size_t)seq * LSEQ;

  // ---- stage A (17 real rows + zero pad to 32) into packed fragments ----
  for (int i = tid; i < 32 * CMOD; i += 256) {
    int r = i >> 8, c = i & 255;
    _Float16 v = (r < LSEQ) ? xin16[(tokbase + r) * CMOD + c] : (_Float16)0.f;
    int tile = (r >> 4) * 8 + (c >> 5);
    sAp[a_pack_idx(tile, r, c & 31)] = v;
  }
  __syncthreads();

  // ---- in-proj GEMM: zx[17,1296] = A(32x256) * Win(256x1296) ----
  for (int nt = wave; nt < DPROJ / 16; nt += 8) {  // 81 n-tiles over 8 waves
    int n0 = nt * 16;
    v8f acc0 = {}, acc1 = {};
#pragma unroll
    for (int kk = 0; kk < 8; ++kk) {
      v16h bb = frag16(Winp, nt * 8 + kk);
      v16h a0 = frag16(sAp, kk);
      v16h a1 = frag16(sAp, 8 + kk);
      acc0 = wmma_f16(a0, bb, acc0);
      acc1 = wmma_f16(a1, bb, acc1);
    }
    store_acc_lds(zx, MAM_ZLD, 0, n0, acc0, LSEQ);
    store_acc_lds(zx, MAM_ZLD, 16, n0, acc1, LSEQ);
  }
  __syncthreads();

  // ---- depthwise conv (taps=4, causal) + SiLU, in place over cols 512..1279 ----
  for (int c = tid; c < CONVD; c += 256) {
    float v[LSEQ];
#pragma unroll
    for (int t = 0; t < LSEQ; ++t) v[t] = zx[t * MAM_ZLD + DIN + c];
    float w0 = cw[c * 4 + 0], w1 = cw[c * 4 + 1], w2 = cw[c * 4 + 2],
          w3 = cw[c * 4 + 3];
    float bc = cb[c];
#pragma unroll
    for (int t = 0; t < LSEQ; ++t) {
      float acc = bc + v[t] * w3;
      if (t >= 1) acc += v[t - 1] * w2;
      if (t >= 2) acc += v[t - 2] * w1;
      if (t >= 3) acc += v[t - 3] * w0;
      zx[t * MAM_ZLD + DIN + c] = silu_f(acc);
    }
  }
  // ---- dt: softplus + cumulative decay (cols 1280..1295, untouched by conv) ----
  if (tid < NH) {
    int h = tid;
    float A = -__expf(Alog[h]);
    float db = dtb[h];
    float run = 0.f;
    for (int t = 0; t < LSEQ; ++t) {
      float xv = zx[t * MAM_ZLD + (DIN + CONVD) + h] + db;
      float sp = (xv > 20.f) ? xv : log1pf(__expf(xv));
      dtv[t * NH + h] = sp;
      run += sp * A;
      cums[t * NH + h] = run;
    }
  }
  __syncthreads();

  // ---- CB[t,s] = Cc[t,:] . Bc[s,:] ----
  for (int p = tid; p < LSEQ * LSEQ; p += 256) {
    int t = p / LSEQ, s = p % LSEQ;
    float acc = 0.f;
    for (int n = 0; n < DSTATE; ++n)
      acc += zx[t * MAM_ZLD + (DIN + DIN + DSTATE) + n] *
             zx[s * MAM_ZLD + (DIN + DIN) + n];
    CBs[p] = acc;
  }
  __syncthreads();

  // ---- M[t,s,h] = CB * exp(cum_t - cum_s) * dt_s  (causal) ----
  for (int i = tid; i < LSEQ * LSEQ * NH; i += 256) {
    int h = i & 15;
    int ts = i >> 4;
    int t = ts / LSEQ, s = ts % LSEQ;
    float m = 0.f;
    if (s <= t)
      m = CBs[t * LSEQ + s] * __expf(cums[t * NH + h] - cums[s * NH + h]) *
          dtv[s * NH + h];
    Mb[i] = m;
  }
  __syncthreads();

  // ---- y[t,d] = sum_s M * xh + D*xh ; gate with silu(z) ----
  for (int d = tid; d < DIN; d += 256) {
    int h = d >> 5;
    float Dh = Dp[h];
    for (int t = 0; t < LSEQ; ++t) {
      float acc = Dh * zx[t * MAM_ZLD + DIN + d];
      for (int s = 0; s <= t; ++s)
        acc += Mb[(t * LSEQ + s) * NH + h] * zx[s * MAM_ZLD + DIN + d];
      float z = zx[t * MAM_ZLD + d];
      yb[t * DIN + d] = acc * silu_f(z);
    }
  }
  __syncthreads();

  // ---- RMS norm over 512 + weight, write f16 ----
  for (int t = wave; t < LSEQ; t += 8) {
    float ss = 0.f;
    for (int d = lane; d < DIN; d += 32) {
      float v = yb[t * DIN + d];
      ss += v * v;
    }
#pragma unroll
    for (int o = 16; o > 0; o >>= 1) ss += __shfl_xor(ss, o, 32);
    float rs = rsqrtf(ss / DIN + 1e-5f);
    for (int d = lane; d < DIN; d += 32)
      yout[(tokbase + t) * DIN + d] = (_Float16)(yb[t * DIN + d] * rs * nw[d]);
  }
}

// ---------------- out-proj: xsum = x + yA*WA + yB*WB ----------------
__global__ __launch_bounds__(256, 1) void k_outproj(
    const float* __restrict__ x,
    const _Float16* __restrict__ yA, const _Float16* __restrict__ yB,
    const _Float16* __restrict__ WAp,  // packed, ktiles=16, ntiles=16
    const _Float16* __restrict__ WBp,
    float* __restrict__ xsum, int T) {
  __shared__ _Float16 sAp[4 * 32 * 16];  // 4 row-tiles, packed fragments
  const int m0 = blockIdx.x * 64;
  const int n0g = blockIdx.y * 64;
  const int wave = threadIdx.x >> 5;
  const int lane = threadIdx.x & 31;
  const int rt = wave & 3;                  // 4 waves over M (16 rows each)
  const int ncol0 = n0g + (wave >> 2) * 32; // 2 waves over N (32 cols each)
  const int nt0 = ncol0 >> 4;
  v8f acc[2] = {{}, {}};

  for (int src = 0; src < 2; ++src) {
    const _Float16* Y = src ? yB : yA;
    const _Float16* W = src ? WBp : WAp;
    for (int kk = 0; kk < DIN / 32; ++kk) {
      __syncthreads();
      for (int i = threadIdx.x; i < 64 * 32; i += 256) {
        int r = i >> 5, c = i & 31;
        int gr = m0 + r;
        _Float16 v =
            (gr < T) ? Y[(size_t)gr * DIN + kk * 32 + c] : (_Float16)0.f;
        sAp[a_pack_idx(r >> 4, r, c)] = v;
      }
      __syncthreads();
      v16h a = frag16(sAp, rt);
      v16h b0 = frag16(W, nt0 * 16 + kk);
      v16h b1 = frag16(W, (nt0 + 1) * 16 + kk);
      acc[0] = wmma_f16(a, b0, acc[0]);
      acc[1] = wmma_f16(a, b1, acc[1]);
    }
  }
#pragma unroll
  for (int half = 0; half < 2; ++half) {
    int n = ncol0 + half * 16 + (lane & 15);
    int r0 = m0 + rt * 16 + ((lane >> 4) << 3);
#pragma unroll
    for (int v = 0; v < 8; ++v) {
      int r = r0 + v;
      if (r < T)
        xsum[(size_t)r * CMOD + n] = x[(size_t)r * CMOD + n] + acc[half][v];
    }
  }
}

// ---------------- fused LN2 + MLP (gelu) + residual ----------------
#define MLP_XS_F (32 * CMOD)
#define MLP_A_H  (2 * 8 * 32 * 16)    // packed A frags (32 rows, K=256)
#define MLP_H_H  (2 * 32 * 32 * 16)   // packed A frags (32 rows, K=1024)
#define MLP_SMEM_BYTES (MLP_XS_F * 4 + MLP_A_H * 2 + MLP_H_H * 2)

__global__ __launch_bounds__(256, 1) void k_mlp(
    const float* __restrict__ xsum,
    const float* __restrict__ n2w, const float* __restrict__ n2b,
    const _Float16* __restrict__ W1p,  // packed, ktiles=8, ntiles=64
    const float* __restrict__ b1,
    const _Float16* __restrict__ W2p,  // packed, ktiles=32, ntiles=16
    const float* __restrict__ b2,
    float* __restrict__ out, int T) {
  extern __shared__ char smem[];
  float* xs = (float*)smem;
  _Float16* a16p = (_Float16*)(smem + MLP_XS_F * 4);
  _Float16* h16p = (_Float16*)(smem + MLP_XS_F * 4 + MLP_A_H * 2);

  const int m0 = blockIdx.x * 32;
  const int tid = threadIdx.x;
  const int wave = tid >> 5;
  const int lane = tid & 31;

  for (int i = tid; i < 32 * CMOD; i += 256) {
    int r = i >> 8, c = i & 255;
    int gr = m0 + r;
    xs[r * CMOD + c] = (gr < T) ? xsum[(size_t)gr * CMOD + c] : 0.f;
  }
  __syncthreads();

  // LN per row -> packed A fragments
  for (int r = wave; r < 32; r += 8) {
    float s1 = 0.f, s2 = 0.f;
    for (int c = lane; c < CMOD; c += 32) {
      float vv = xs[r * CMOD + c];
      s1 += vv;
      s2 += vv * vv;
    }
#pragma unroll
    for (int o = 16; o > 0; o >>= 1) {
      s1 += __shfl_xor(s1, o, 32);
      s2 += __shfl_xor(s2, o, 32);
    }
    float m = s1 / CMOD;
    float rs = rsqrtf(s2 / CMOD - m * m + 1e-5f);
    for (int c = lane; c < CMOD; c += 32) {
      float v = (xs[r * CMOD + c] - m) * rs * n2w[c] + n2b[c];
      int tile = (r >> 4) * 8 + (c >> 5);
      a16p[a_pack_idx(tile, r, c & 31)] = (_Float16)v;
    }
  }
  __syncthreads();

  // GEMM1: hidden = gelu(a @ W1 + b1) -> h16p (packed for GEMM2)
  for (int nt = wave; nt < HID / 16; nt += 8) {
    int n0 = nt * 16;
    v8f acc0 = {}, acc1 = {};
#pragma unroll
    for (int kk = 0; kk < 8; ++kk) {
      v16h bb = frag16(W1p, nt * 8 + kk);
      v16h a0 = frag16(a16p, kk);
      v16h a1 = frag16(a16p, 8 + kk);
      acc0 = wmma_f16(a0, bb, acc0);
      acc1 = wmma_f16(a1, bb, acc1);
    }
    int n = n0 + (lane & 15);
    float b1v = b1[n];
    int rsub = (lane >> 4) << 3;
    int kk2 = n >> 5, kp2 = n & 31;  // position of column n in GEMM2's K dim
#pragma unroll
    for (int v = 0; v < 8; ++v) {
      float x0 = acc0[v] + b1v;
      float g0 = 0.5f * x0 * (1.f + erff(x0 * 0.70710678f));
      h16p[a_pack_idx(0 * 32 + kk2, rsub + v, kp2)] = (_Float16)g0;
      float x1 = acc1[v] + b1v;
      float g1 = 0.5f * x1 * (1.f + erff(x1 * 0.70710678f));
      h16p[a_pack_idx(1 * 32 + kk2, 16 + rsub + v, kp2)] = (_Float16)g1;
    }
  }
  __syncthreads();

  // GEMM2: out = xs + h @ W2 + b2
  for (int nt = wave; nt < CMOD / 16; nt += 8) {
    int n0 = nt * 16;
    v8f acc0 = {}, acc1 = {};
    for (int kk = 0; kk < HID / 32; ++kk) {
      v16h bb = frag16(W2p, nt * 32 + kk);
      v16h a0 = frag16(h16p, kk);
      v16h a1 = frag16(h16p, 32 + kk);
      acc0 = wmma_f16(a0, bb, acc0);
      acc1 = wmma_f16(a1, bb, acc1);
    }
    int n = n0 + (lane & 15);
    float b2v = b2[n];
    int rsub = (lane >> 4) << 3;
#pragma unroll
    for (int v = 0; v < 8; ++v) {
      int r = rsub + v;
      int gr = m0 + r;
      if (gr < T) out[(size_t)gr * CMOD + n] = xs[r * CMOD + n] + b2v + acc0[v];
      int r1 = 16 + rsub + v;
      int gr1 = m0 + r1;
      if (gr1 < T)
        out[(size_t)gr1 * CMOD + n] = xs[r1 * CMOD + n] + b2v + acc1[v];
    }
  }
}

// ---------------- launch ----------------
extern "C" void kernel_launch(void* const* d_in, const int* in_sizes, int n_in,
                              void* d_out, int out_size, void* d_ws, size_t ws_size,
                              hipStream_t stream) {
  (void)in_sizes; (void)n_in; (void)out_size; (void)ws_size;
  const float* x     = (const float*)d_in[0];
  const float* n1w   = (const float*)d_in[1];
  const float* n1b   = (const float*)d_in[2];
  const float* n2w   = (const float*)d_in[3];
  const float* n2b   = (const float*)d_in[4];
  const float* w1    = (const float*)d_in[5];
  const float* b1    = (const float*)d_in[6];
  const float* w2    = (const float*)d_in[7];
  const float* b2    = (const float*)d_in[8];
  const float* a_in  = (const float*)d_in[9];
  const float* a_cw  = (const float*)d_in[10];
  const float* a_cb  = (const float*)d_in[11];
  const float* a_dtb = (const float*)d_in[12];
  const float* a_Alog= (const float*)d_in[13];
  const float* a_D   = (const float*)d_in[14];
  const float* a_nw  = (const float*)d_in[15];
  const float* a_out = (const float*)d_in[16];
  const float* i_in  = (const float*)d_in[17];
  const float* i_cw  = (const float*)d_in[18];
  const float* i_cb  = (const float*)d_in[19];
  const float* i_dtb = (const float*)d_in[20];
  const float* i_Alog= (const float*)d_in[21];
  const float* i_D   = (const float*)d_in[22];
  const float* i_nw  = (const float*)d_in[23];
  const float* i_out = (const float*)d_in[24];

  char* ws = (char*)d_ws;
  size_t off = 0;
  auto alloc = [&](size_t bytes) -> char* {
    char* p = ws + off;
    off += (bytes + 255) & ~(size_t)255;
    return p;
  };
  _Float16* WaIn  = (_Float16*)alloc((size_t)CMOD * DPROJ * 2);
  _Float16* WiIn  = (_Float16*)alloc((size_t)CMOD * DPROJ * 2);
  _Float16* WaOut = (_Float16*)alloc((size_t)DIN * CMOD * 2);
  _Float16* WiOut = (_Float16*)alloc((size_t)DIN * CMOD * 2);
  _Float16* W1h   = (_Float16*)alloc((size_t)CMOD * HID * 2);
  _Float16* W2h   = (_Float16*)alloc((size_t)HID * CMOD * 2);
  _Float16* xn16  = (_Float16*)alloc((size_t)TOK * CMOD * 2);
  _Float16* xfl16 = (_Float16*)alloc((size_t)TOK * CMOD * 2);
  _Float16* yA16  = (_Float16*)alloc((size_t)TOK * DIN * 2);
  _Float16* yB16  = (_Float16*)alloc((size_t)TOK * DIN * 2);
  float*    xsum  = (float*)alloc((size_t)TOK * CMOD * 4);

  dim3 blk(256);
  // weight prep (f32 [N][K] -> fragment-packed f16 [K,N] tiles)
  k_wpack<<<(DPROJ * CMOD + 255) / 256, blk, 0, stream>>>(a_in, WaIn, CMOD, DPROJ);
  k_wpack<<<(DPROJ * CMOD + 255) / 256, blk, 0, stream>>>(i_in, WiIn, CMOD, DPROJ);
  k_wpack<<<(CMOD * DIN + 255) / 256, blk, 0, stream>>>(a_out, WaOut, DIN, CMOD);
  k_wpack<<<(CMOD * DIN + 255) / 256, blk, 0, stream>>>(i_out, WiOut, DIN, CMOD);
  k_wpack<<<(HID * CMOD + 255) / 256, blk, 0, stream>>>(w1, W1h, CMOD, HID);
  k_wpack<<<(CMOD * HID + 255) / 256, blk, 0, stream>>>(w2, W2h, HID, CMOD);

  k_ln1<<<TOK, blk, 0, stream>>>(x, n1w, n1b, xn16, xfl16);

  k_mamba<<<SEQ, blk, MAM_SMEM_BYTES, stream>>>(xn16, WaIn, a_cw, a_cb, a_dtb,
                                                a_Alog, a_D, a_nw, yA16);
  k_mamba<<<SEQ, blk, MAM_SMEM_BYTES, stream>>>(xfl16, WiIn, i_cw, i_cb, i_dtb,
                                                i_Alog, i_D, i_nw, yB16);

  k_outproj<<<dim3((TOK + 63) / 64, CMOD / 64), blk, 0, stream>>>(
      x, yA16, yB16, WaOut, WiOut, xsum, TOK);

  k_mlp<<<(TOK + 31) / 32, blk, MLP_SMEM_BYTES, stream>>>(
      xsum, n2w, n2b, W1h, b1, W2h, b2, (float*)d_out, TOK);
}
